// AKT_1872605741871
// MI455X (gfx1250) — matrix operand order repaired and checked by reading, not compiled, not run on hardware
//
#include <hip/hip_runtime.h>
#include <math.h>

// ---------------------------------------------------------------------------
// Types for CDNA5 WMMA (wave32, 16x16x32 bf16 -> f32)
// ---------------------------------------------------------------------------
typedef __attribute__((ext_vector_type(16))) __bf16 v16bf;
typedef __attribute__((ext_vector_type(8)))  float  v8f;

union FragBF {
    v16bf v;
    uint4 q[2];
};

__device__ __forceinline__ unsigned short bf16r(float f) {
    union { float f; unsigned int u; } v;
    v.f = f;
    unsigned int r = v.u + 0x7FFFu + ((v.u >> 16) & 1u);  // round-to-nearest-even
    return (unsigned short)(r >> 16);
}

__device__ __forceinline__ unsigned int pack2(float lo, float hi) {
    return (unsigned int)bf16r(lo) | ((unsigned int)bf16r(hi) << 16);
}

__device__ __forceinline__ float gelu_exact(float x) {
    return 0.5f * x * (1.0f + erff(x * 0.7071067811865476f));
}

// Async global->LDS copy of 16 bytes (gfx1250 GLOBAL_LOAD_ASYNC_TO_LDS_B128).
// ldsaddr: wave-relative LDS byte address; gaddr: 64-bit global address.
__device__ __forceinline__ void async_b128_to_lds(unsigned ldsaddr, unsigned long long gaddr) {
    asm volatile("global_load_async_to_lds_b128 %0, %1, off"
                 :: "v"(ldsaddr), "v"(gaddr)
                 : "memory");
}

__device__ __forceinline__ void wait_async0() {
    asm volatile("s_wait_asynccnt 0" ::: "memory");
}

// ---------------------------------------------------------------------------
// Model constants
// ---------------------------------------------------------------------------
#define BB 32
#define TT 1024
#define DM 256
#define NH 8
#define DH 32
#define ROWS (BB * TT)   // 32768

// ---------------------------------------------------------------------------
// Weight transpose + f32->bf16 convert: W[K,N] -> Wt[N,K] (bf16)
// ---------------------------------------------------------------------------
__global__ void k_convT(const float* __restrict__ W, unsigned short* __restrict__ Wt,
                        int K, int N) {
    int i = blockIdx.x * 256 + threadIdx.x;
    if (i >= N * K) return;
    int n = i / K, k = i - n * K;
    Wt[i] = bf16r(W[(size_t)k * N + n]);
}

// ---------------------------------------------------------------------------
// Embedding + input projection + difficulty + sinusoidal PE
// grid = ROWS blocks, 256 threads
// ---------------------------------------------------------------------------
__global__ __launch_bounds__(256)
void k_embed(const float* __restrict__ X, const float* __restrict__ tagE,
             const float* __restrict__ partE, const float* __restrict__ confE,
             const float* __restrict__ projW, const float* __restrict__ projB,
             const float* __restrict__ diffT, float* __restrict__ H) {
    __shared__ float cmb[101];
    __shared__ float sdiff;
    const int r = blockIdx.x;
    const int tid = threadIdx.x;
    const float* xr = X + (size_t)r * 14;

    int ids[7];
#pragma unroll
    for (int j = 0; j < 7; ++j) {
        int id = (int)xr[j];
        id = id < 0 ? 0 : (id > 300 ? 300 : id);
        ids[j] = id;
    }

    if (tid < 64) {
        int cnt = 0; float s = 0.f;
#pragma unroll
        for (int j = 0; j < 7; ++j) {
            if (ids[j] > 0) cnt++;
            s += tagE[ids[j] * 64 + tid];
        }
        cmb[tid] = s / (float)(cnt > 0 ? cnt : 1);
    } else if (tid < 69) {
        const int map5[5] = {7, 8, 9, 12, 13};   // correct, elapsed, changed, steps, cum
        cmb[tid] = xr[map5[tid - 64]];
    } else if (tid < 85) {
        int pid = (int)xr[10]; pid = pid < 0 ? 0 : (pid > 7 ? 7 : pid);
        cmb[tid] = partE[pid * 16 + (tid - 69)];
    } else if (tid < 101) {
        int cid = (int)xr[11]; cid = cid < 0 ? 0 : (cid > 4 ? 4 : cid);
        cmb[tid] = confE[cid * 16 + (tid - 85)];
    } else if (tid == 101) {
        float s = 0.f;
#pragma unroll
        for (int j = 0; j < 7; ++j) s += diffT[ids[j]];
        sdiff = s * (1.0f / 7.0f);
    }
    __syncthreads();

    float a = projB[tid] + sdiff;
    for (int k = 0; k < 101; ++k) a += cmb[k] * projW[k * 256 + tid];

    const int t = r & (TT - 1);
    const int i2 = (tid >> 1) * 2;
    float fr = __expf((float)i2 * (-9.210340371976184f / 256.0f));  // -ln(10000)/256
    float ang = (float)t * fr;
    a += (tid & 1) ? __cosf(ang) : __sinf(ang);

    H[(size_t)r * DM + tid] = a;
}

// ---------------------------------------------------------------------------
// Tiled bf16 WMMA GEMM: C[M,N] = act(A[M,K] @ W[K,N] + bias)
// A: f32 (converted to bf16 during LDS staging), Wt: bf16 [N,K] (pre-transposed,
// staged with async global->LDS DMA, no VGPR round-trip).
// Block tile 128x64x32, 8 waves (wave32), each wave 32x32 via 2x2 WMMA tiles.
// act: 0 = none, 1 = exact GELU
// ---------------------------------------------------------------------------
#define GBM 128
#define GBN 64
#define GBK 32

__global__ __launch_bounds__(256)
void k_gemm(const float* __restrict__ A, const unsigned short* __restrict__ Wt,
            const float* __restrict__ bias, float* __restrict__ C,
            int M, int N, int K, int act) {
    __shared__ __align__(16) unsigned short As[GBM][GBK];
    __shared__ __align__(16) unsigned short Bs[GBN][GBK];

    const int tid  = threadIdx.x;
    const int lane = tid & 31;
    const int wid  = tid >> 5;
    const int wm   = wid & 3;        // 0..3 -> M offset wm*32
    const int wn   = wid >> 2;       // 0..1 -> N offset wn*32
    const int lo   = lane & 15;
    const int hi   = lane >> 4;

    const int m0 = blockIdx.y * GBM;
    const int n0 = blockIdx.x * GBN;

    v8f acc[2][2];
#pragma unroll
    for (int mt = 0; mt < 2; ++mt)
#pragma unroll
        for (int nt = 0; nt < 2; ++nt)
#pragma unroll
            for (int p = 0; p < 8; ++p) acc[mt][nt][p] = 0.0f;

    const int arow = tid >> 1;          // 0..127
    const int aseg = (tid & 1) * 16;    // 0 or 16
    const int brow = tid >> 2;          // 0..63
    const int bseg = (tid & 3) * 8;     // 0,8,16,24

    const unsigned ldsB = (unsigned)(size_t)&Bs[brow][bseg];
    const unsigned short* gB0 = Wt + (size_t)(n0 + brow) * K + bseg;

    for (int kt = 0; kt < K; kt += GBK) {
        // Stage B tile: async DMA straight into LDS (bf16 already in [N,K]).
        async_b128_to_lds(ldsB, (unsigned long long)(size_t)(gB0 + kt));

        // Stage A tile (f32 -> bf16) through VGPRs (fused conversion).
        {
            const float4* ap = (const float4*)(A + (size_t)(m0 + arow) * K + kt + aseg);
            unsigned int pk[8];
#pragma unroll
            for (int q = 0; q < 4; ++q) {
                float4 f = ap[q];
                pk[q * 2]     = pack2(f.x, f.y);
                pk[q * 2 + 1] = pack2(f.z, f.w);
            }
            uint4* dst = (uint4*)&As[arow][aseg];
            dst[0] = make_uint4(pk[0], pk[1], pk[2], pk[3]);
            dst[1] = make_uint4(pk[4], pk[5], pk[6], pk[7]);
        }
        wait_async0();          // this wave's async B-tile bytes are in LDS
        __syncthreads();        // publish A+B tiles to all waves

        FragBF af[2], bf[2];
#pragma unroll
        for (int mt = 0; mt < 2; ++mt) {
            int mm = wm * 32 + mt * 16 + lo;
            af[mt].q[0] = *(const uint4*)&As[mm][hi * 8];
            af[mt].q[1] = *(const uint4*)&As[mm][hi * 8 + 16];
        }
#pragma unroll
        for (int nt = 0; nt < 2; ++nt) {
            int nn = wn * 32 + nt * 16 + lo;
            bf[nt].q[0] = *(const uint4*)&Bs[nn][hi * 8];
            bf[nt].q[1] = *(const uint4*)&Bs[nn][hi * 8 + 16];
        }
#pragma unroll
        for (int mt = 0; mt < 2; ++mt)
#pragma unroll
            for (int nt = 0; nt < 2; ++nt)
                acc[mt][nt] = __builtin_amdgcn_wmma_f32_16x16x32_bf16(
                    false, af[mt].v, false, bf[nt].v, (short)0, acc[mt][nt], false, false);
        __syncthreads();
    }

    // Epilogue: C-layout row = p + hi*8, col = lo
#pragma unroll
    for (int mt = 0; mt < 2; ++mt)
#pragma unroll
        for (int nt = 0; nt < 2; ++nt) {
            int cbase = n0 + wn * 32 + nt * 16 + lo;
            float bv = bias[cbase];
#pragma unroll
            for (int p = 0; p < 8; ++p) {
                int r = m0 + wm * 32 + mt * 16 + hi * 8 + p;
                float v = acc[mt][nt][p] + bv;
                if (act == 1) v = gelu_exact(v);
                C[(size_t)r * N + cbase] = v;
            }
        }
}

// ---------------------------------------------------------------------------
// Flash-style AKT attention (causal + distance decay), bf16 WMMA.
// grid = (T/64, NH, B), block = 128 threads (4 waves), one 16-query tile/wave.
// ---------------------------------------------------------------------------
__global__ __launch_bounds__(128)
void k_attn(const float* __restrict__ Q, const float* __restrict__ Kb,
            const float* __restrict__ V, const float* __restrict__ gammaP,
            float* __restrict__ O) {
    __shared__ __align__(16) unsigned short Qs[64][32];
    __shared__ __align__(16) unsigned short Ks[32][32];
    __shared__ __align__(16) unsigned short Vt[32][32];
    __shared__ __align__(16) unsigned short Ps[4][16][32];

    const int tid  = threadIdx.x;
    const int lane = tid & 31;
    const int w    = tid >> 5;       // wave 0..3
    const int lo   = lane & 15;
    const int hi   = lane >> 4;

    const int qt = blockIdx.x;       // 0..15
    const int h  = blockIdx.y;
    const int b  = blockIdx.z;
    const int qbase = qt * 64;

    // Stage Q tile (64x32 f32 -> bf16)
    {
        int r = tid >> 1, seg = (tid & 1) * 16;
        const float4* p = (const float4*)(Q + ((size_t)b * TT + qbase + r) * DM + h * DH + seg);
        unsigned int pk[8];
#pragma unroll
        for (int q = 0; q < 4; ++q) {
            float4 f = p[q];
            pk[q * 2]     = pack2(f.x, f.y);
            pk[q * 2 + 1] = pack2(f.z, f.w);
        }
        uint4* dst = (uint4*)&Qs[r][seg];
        dst[0] = make_uint4(pk[0], pk[1], pk[2], pk[3]);
        dst[1] = make_uint4(pk[4], pk[5], pk[6], pk[7]);
    }
    __syncthreads();

    FragBF qf;
    {
        int qrow = w * 16 + lo;
        qf.q[0] = *(const uint4*)&Qs[qrow][hi * 8];
        qf.q[1] = *(const uint4*)&Qs[qrow][hi * 8 + 16];
    }

    const float gmm   = logf(1.0f + __expf(gammaP[h]));      // softplus
    const float scale = 0.17677669529663687f;                // 1/sqrt(32)

    float mrow[8], lrow[8];
    v8f acc0, acc1;
#pragma unroll
    for (int p = 0; p < 8; ++p) { mrow[p] = -3.0e38f; lrow[p] = 0.f; acc0[p] = 0.f; acc1[p] = 0.f; }

    const int ntile  = qt * 2 + 2;       // causal bound: s-tiles covering [0, qbase+63]
    const int tqbase = qbase + w * 16;

    for (int st = 0; st < ntile; ++st) {
        const int sb = st * 32;
        // Stage K tile and V^T tile
        {
            int r = tid >> 2, seg = (tid & 3) * 8;
            const float4* kp = (const float4*)(Kb + ((size_t)b * TT + sb + r) * DM + h * DH + seg);
            float4 f0 = kp[0], f1 = kp[1];
            uint4* kd = (uint4*)&Ks[r][seg];
            kd[0] = make_uint4(pack2(f0.x, f0.y), pack2(f0.z, f0.w),
                               pack2(f1.x, f1.y), pack2(f1.z, f1.w));
            const float4* vp = (const float4*)(V + ((size_t)b * TT + sb + r) * DM + h * DH + seg);
            float4 g0 = vp[0], g1 = vp[1];
            Vt[seg + 0][r] = bf16r(g0.x); Vt[seg + 1][r] = bf16r(g0.y);
            Vt[seg + 2][r] = bf16r(g0.z); Vt[seg + 3][r] = bf16r(g0.w);
            Vt[seg + 4][r] = bf16r(g1.x); Vt[seg + 5][r] = bf16r(g1.y);
            Vt[seg + 6][r] = bf16r(g1.z); Vt[seg + 7][r] = bf16r(g1.w);
        }
        __syncthreads();

        // S = Q @ K^T  (two 16-col s sub-tiles)
        v8f z8 = {0.f, 0.f, 0.f, 0.f, 0.f, 0.f, 0.f, 0.f};
        FragBF kf0, kf1;
        kf0.q[0] = *(const uint4*)&Ks[lo][hi * 8];
        kf0.q[1] = *(const uint4*)&Ks[lo][hi * 8 + 16];
        kf1.q[0] = *(const uint4*)&Ks[16 + lo][hi * 8];
        kf1.q[1] = *(const uint4*)&Ks[16 + lo][hi * 8 + 16];
        v8f S0 = __builtin_amdgcn_wmma_f32_16x16x32_bf16(false, qf.v, false, kf0.v, (short)0, z8, false, false);
        v8f S1 = __builtin_amdgcn_wmma_f32_16x16x32_bf16(false, qf.v, false, kf1.v, (short)0, z8, false, false);

        // mask + decay + online softmax
#pragma unroll
        for (int p = 0; p < 8; ++p) {
            const int tq = tqbase + p + hi * 8;
            const int s0 = sb + lo;
            const int s1 = sb + 16 + lo;
            float sv0 = (s0 > tq) ? -3.0e38f : (S0[p] * scale - gmm * (float)(tq - s0));
            float sv1 = (s1 > tq) ? -3.0e38f : (S1[p] * scale - gmm * (float)(tq - s1));

            float rm = fmaxf(sv0, sv1);
#pragma unroll
            for (int d = 1; d <= 8; d <<= 1) rm = fmaxf(rm, __shfl_xor(rm, d, 32));
            float nm = fmaxf(mrow[p], rm);
            float cf = __expf(mrow[p] - nm);
            float e0 = (sv0 <= -1.0e37f) ? 0.f : __expf(sv0 - nm);
            float e1 = (sv1 <= -1.0e37f) ? 0.f : __expf(sv1 - nm);
            float es = e0 + e1;
#pragma unroll
            for (int d = 1; d <= 8; d <<= 1) es += __shfl_xor(es, d, 32);

            lrow[p] = lrow[p] * cf + es;
            acc0[p] *= cf;
            acc1[p] *= cf;
            mrow[p] = nm;

            Ps[w][p + hi * 8][lo]      = bf16r(e0);
            Ps[w][p + hi * 8][16 + lo] = bf16r(e1);
        }
        __syncthreads();

        // O += P @ V  (two 16-col d sub-tiles)
        FragBF pf, vf0, vf1;
        pf.q[0]  = *(const uint4*)&Ps[w][lo][hi * 8];
        pf.q[1]  = *(const uint4*)&Ps[w][lo][hi * 8 + 16];
        vf0.q[0] = *(const uint4*)&Vt[lo][hi * 8];
        vf0.q[1] = *(const uint4*)&Vt[lo][hi * 8 + 16];
        vf1.q[0] = *(const uint4*)&Vt[16 + lo][hi * 8];
        vf1.q[1] = *(const uint4*)&Vt[16 + lo][hi * 8 + 16];
        acc0 = __builtin_amdgcn_wmma_f32_16x16x32_bf16(false, pf.v, false, vf0.v, (short)0, acc0, false, false);
        acc1 = __builtin_amdgcn_wmma_f32_16x16x32_bf16(false, pf.v, false, vf1.v, (short)0, acc1, false, false);
        __syncthreads();
    }

#pragma unroll
    for (int p = 0; p < 8; ++p) {
        const int tq = tqbase + p + hi * 8;
        const float inv = 1.0f / lrow[p];
        size_t base = ((size_t)b * TT + tq) * DM + h * DH;
        O[base + lo]      = acc0[p] * inv;
        O[base + 16 + lo] = acc1[p] * inv;
    }
}

// ---------------------------------------------------------------------------
// Residual add + LayerNorm (in place on H): H = LN(H + R) * g + b
// grid = ROWS, block = 256
// ---------------------------------------------------------------------------
__global__ __launch_bounds__(256)
void k_lnres(float* __restrict__ H, const float* __restrict__ R,
             const float* __restrict__ g, const float* __restrict__ bta) {
    const int r = blockIdx.x, c = threadIdx.x;
    const size_t idx = (size_t)r * DM + c;
    float v = H[idx] + R[idx];
    float s = v, s2 = v * v;
#pragma unroll
    for (int d = 1; d < 32; d <<= 1) {
        s  += __shfl_xor(s, d, 32);
        s2 += __shfl_xor(s2, d, 32);
    }
    __shared__ float as1[8], as2[8];
    const int lane = c & 31, wid = c >> 5;
    if (lane == 0) { as1[wid] = s; as2[wid] = s2; }
    __syncthreads();
    float ts = 0.f, ts2 = 0.f;
#pragma unroll
    for (int i = 0; i < 8; ++i) { ts += as1[i]; ts2 += as2[i]; }
    float mean = ts * (1.0f / DM);
    float var  = ts2 * (1.0f / DM) - mean * mean;
    H[idx] = (v - mean) * rsqrtf(var + 1e-5f) * g[c] + bta[c];
}

// ---------------------------------------------------------------------------
// Mean pool over T: pooled[b, c] = mean_t H[b, t, c]
// ---------------------------------------------------------------------------
__global__ __launch_bounds__(256)
void k_pool(const float* __restrict__ H, float* __restrict__ P) {
    const int b = blockIdx.x, c = threadIdx.x;
    const float* p = H + (size_t)b * TT * DM + c;
    float s = 0.f;
    for (int t = 0; t < TT; ++t) s += p[(size_t)t * DM];
    P[b * DM + c] = s * (1.0f / TT);
}

// ---------------------------------------------------------------------------
// Head: out[b, n] = pooled[b,:] @ head_w[:, n] + head_b[n]   (N=300)
// ---------------------------------------------------------------------------
__global__ __launch_bounds__(320)
void k_head(const float* __restrict__ P, const float* __restrict__ W,
            const float* __restrict__ bs, float* __restrict__ out) {
    __shared__ float pl[DM];
    const int b = blockIdx.x, tid = threadIdx.x;
    if (tid < DM) pl[tid] = P[b * DM + tid];
    __syncthreads();
    if (tid < 300) {
        float s = bs[tid];
        for (int k = 0; k < DM; ++k) s += pl[k] * W[k * 300 + tid];
        out[b * 300 + tid] = s;
    }
}

// ---------------------------------------------------------------------------
// Host-side orchestration
// ---------------------------------------------------------------------------
extern "C" void kernel_launch(void* const* d_in, const int* in_sizes, int n_in,
                              void* d_out, int out_size, void* d_ws, size_t ws_size,
                              hipStream_t stream) {
    (void)in_sizes; (void)n_in; (void)out_size; (void)ws_size;

    const float* x      = (const float*)d_in[0];
    const float* tagE   = (const float*)d_in[1];
    const float* partE  = (const float*)d_in[2];
    const float* confE  = (const float*)d_in[3];
    const float* projW  = (const float*)d_in[4];
    const float* projB  = (const float*)d_in[5];
    const float* diffT  = (const float*)d_in[6];
    auto L = [&](int i, int j) { return (const float*)d_in[7 + i * 17 + j]; };
    const float* headW  = (const float*)d_in[75];
    const float* headB  = (const float*)d_in[76];

    // Workspace carve
    char* ws = (char*)d_ws;
    size_t off = 0;
    auto carve = [&](size_t bytes) -> void* {
        void* p = ws + off;
        off += (bytes + 255) & ~(size_t)255;
        return p;
    };

    // Per-layer transposed bf16 weights: wq,wk,wv,wo (256x256), w1 (256x1024), w2 (1024x256)
    struct WDim { int j, K, N; };
    const WDim wd[6] = {{0,256,256},{2,256,256},{4,256,256},{6,256,256},{11,256,1024},{13,1024,256}};
    unsigned short* wT[4][6];
    for (int i = 0; i < 4; ++i)
        for (int k = 0; k < 6; ++k)
            wT[i][k] = (unsigned short*)carve((size_t)wd[k].K * wd[k].N * 2);

    float* hbuf   = (float*)carve((size_t)ROWS * DM * 4);
    float* qbuf   = (float*)carve((size_t)ROWS * DM * 4);
    float* kbuf   = (float*)carve((size_t)ROWS * DM * 4);
    float* vbuf   = (float*)carve((size_t)ROWS * DM * 4);
    float* abuf   = (float*)carve((size_t)ROWS * DM * 4);
    float* ffbuf  = (float*)carve((size_t)ROWS * 1024 * 4);
    float* pooled = (float*)carve((size_t)BB * DM * 4);

    // 1) Weight conversion (transpose + bf16)
    for (int i = 0; i < 4; ++i)
        for (int k = 0; k < 6; ++k) {
            int tot = wd[k].K * wd[k].N;
            k_convT<<<(tot + 255) / 256, 256, 0, stream>>>(L(i, wd[k].j), wT[i][k], wd[k].K, wd[k].N);
        }

    // 2) Embedding + projection + PE
    k_embed<<<ROWS, 256, 0, stream>>>(x, tagE, partE, confE, projW, projB, diffT, hbuf);

    // 3) Transformer layers
    const dim3 gP(DM / GBN, ROWS / GBM);        // 256-wide GEMMs
    const dim3 gF1(1024 / GBN, ROWS / GBM);     // FFN up
    for (int i = 0; i < 4; ++i) {
        k_gemm<<<gP, 256, 0, stream>>>(hbuf, wT[i][0], L(i, 1), qbuf, ROWS, DM, DM, 0);
        k_gemm<<<gP, 256, 0, stream>>>(hbuf, wT[i][1], L(i, 3), kbuf, ROWS, DM, DM, 0);
        k_gemm<<<gP, 256, 0, stream>>>(hbuf, wT[i][2], L(i, 5), vbuf, ROWS, DM, DM, 0);

        k_attn<<<dim3(TT / 64, NH, BB), 128, 0, stream>>>(qbuf, kbuf, vbuf, L(i, 8), abuf);

        k_gemm<<<gP, 256, 0, stream>>>(abuf, wT[i][3], L(i, 7), qbuf, ROWS, DM, DM, 0);
        k_lnres<<<ROWS, 256, 0, stream>>>(hbuf, qbuf, L(i, 9), L(i, 10));

        k_gemm<<<gF1, 256, 0, stream>>>(hbuf, wT[i][4], L(i, 12), ffbuf, ROWS, 1024, DM, 1);
        k_gemm<<<gP, 256, 0, stream>>>(ffbuf, wT[i][5], L(i, 14), qbuf, ROWS, DM, 1024, 0);
        k_lnres<<<ROWS, 256, 0, stream>>>(hbuf, qbuf, L(i, 15), L(i, 16));
    }

    // 4) Pool + head
    k_pool<<<BB, 256, 0, stream>>>(hbuf, pooled);
    k_head<<<BB, 320, 0, stream>>>(pooled, headW, headB, (float*)d_out);
}
